// Attention_68006512164916
// MI455X (gfx1250) — matrix operand order repaired and checked by reading, not compiled, not run on hardware
//
#include <hip/hip_runtime.h>

// ---------- types ----------
typedef unsigned short u16;
typedef __attribute__((ext_vector_type(16))) __bf16 bf16x16;
typedef __attribute__((ext_vector_type(8)))  float  f32x8;
typedef __attribute__((ext_vector_type(4)))  float  f32x4;
typedef __attribute__((ext_vector_type(4)))  unsigned u32x4;
typedef __attribute__((ext_vector_type(2)))  unsigned u32x2;

#define WMMA_BF16(a, b, c) \
    __builtin_amdgcn_wmma_f32_16x16x32_bf16(false, (a), false, (b), (short)0, (c), false, false)

// ds_swizzle group-of-32 xor patterns (and_mask=0x1f, or=0, xor at [14:10])
#define SWZF(v, imm) __builtin_bit_cast(float, __builtin_amdgcn_ds_swizzle(__builtin_bit_cast(int, (v)), (imm)))

__device__ __forceinline__ u16 f2bf(float f) {
    unsigned u = __builtin_bit_cast(unsigned, f);
    unsigned r = u + 0x7fffu + ((u >> 16) & 1u);   // RNE
    return (u16)(r >> 16);
}
__device__ __forceinline__ float bf2f(u16 h) {
    return __builtin_bit_cast(float, ((unsigned)h) << 16);
}

// A operand, 16x32 bf16 tile. lane&15 = row M; lanes>=16 use K offset 8.
// per lane: K = koff+{0..7} and koff+16+{0..7}  (two b128 loads)
__device__ __forceinline__ bf16x16 load_a16x32(const u16* p, int row0, int ld, int k0, int lane) {
    const u16* rp = p + (size_t)(row0 + (lane & 15)) * ld + k0 + ((lane >> 4) << 3);
    u32x4 lo = *(const u32x4*)(rp);
    u32x4 hi = *(const u32x4*)(rp + 16);
    return __builtin_bit_cast(bf16x16, __builtin_shufflevector(lo, hi, 0, 1, 2, 3, 4, 5, 6, 7));
}

// B operand, 32x16 bf16 tile. lane&15 = col N; lanes>=16 use K offset 8.
// source row (row0 + lane&15) supplies K = koff+{0..7} and koff+16+{0..7} (two b128 loads).
__device__ __forceinline__ bf16x16 load_b32x16(const u16* p, int row0, int ld, int k0, int lane) {
    const u16* rp = p + (size_t)(row0 + (lane & 15)) * ld + k0 + ((lane >> 4) << 3);
    u32x4 lo = *(const u32x4*)(rp);
    u32x4 hi = *(const u32x4*)(rp + 16);
    return __builtin_bit_cast(bf16x16, __builtin_shufflevector(lo, hi, 0, 1, 2, 3, 4, 5, 6, 7));
}

// ---------- fp32 -> bf16 conversion ----------
__global__ void cvt_kernel(const float* __restrict__ in, u16* __restrict__ out, int n4) {
    int i = blockIdx.x * blockDim.x + threadIdx.x;
    if (i >= n4) return;
    f32x4 v = ((const f32x4*)in)[i];
    unsigned a = (unsigned)f2bf(v[0]) | ((unsigned)f2bf(v[1]) << 16);
    unsigned b = (unsigned)f2bf(v[2]) | ((unsigned)f2bf(v[3]) << 16);
    ((u32x2*)out)[i] = (u32x2){a, b};
}

// ---------- QKV GEMM: (16384x1024) @ (3072x1024)^T, epilogue scatters q/k/v(T) ----------
__global__ __launch_bounds__(256) void gemm_qkv(const u16* __restrict__ A, const u16* __restrict__ W,
                                                u16* __restrict__ Qb, u16* __restrict__ Kb,
                                                u16* __restrict__ VTb) {
    const int lane  = threadIdx.x & 31;
    const int wid   = threadIdx.x >> 5;
    const int mbase = blockIdx.y * 128 + (wid >> 2) * 64;
    const int nbase = blockIdx.x * 128 + (wid & 3) * 32;

    const f32x8 vz = {0.f, 0.f, 0.f, 0.f, 0.f, 0.f, 0.f, 0.f};
    f32x8 acc[4][2];
#pragma unroll
    for (int i = 0; i < 4; ++i) { acc[i][0] = vz; acc[i][1] = vz; }

    for (int k0 = 0; k0 < 1024; k0 += 32) {
        if (k0 + 32 < 1024) {
            __builtin_prefetch(A + (size_t)(mbase + (lane & 15)) * 1024 + k0 + 32, 0, 1);
            __builtin_prefetch(W + (size_t)(nbase + (lane & 15)) * 1024 + k0 + 32, 0, 1);
        }
        bf16x16 av[4];
#pragma unroll
        for (int i = 0; i < 4; ++i) av[i] = load_a16x32(A, mbase + i * 16, 1024, k0, lane);
        bf16x16 bv[2];
#pragma unroll
        for (int j = 0; j < 2; ++j) bv[j] = load_b32x16(W, nbase + j * 16, 1024, k0, lane);
#pragma unroll
        for (int i = 0; i < 4; ++i)
#pragma unroll
            for (int j = 0; j < 2; ++j) acc[i][j] = WMMA_BF16(av[i], bv[j], acc[i][j]);
    }

    const int cc = lane & 15;
    const int rowadd = (lane < 16) ? 0 : 8;
#pragma unroll
    for (int i = 0; i < 4; ++i)
#pragma unroll
        for (int j = 0; j < 2; ++j)
#pragma unroll
            for (int r = 0; r < 8; ++r) {
                float v = acc[i][j][r];
                int m = mbase + i * 16 + r + rowadd;
                int n = nbase + j * 16 + cc;
                int s = n >> 10, c2 = n & 1023, h = c2 >> 6, d = c2 & 63;
                int b = m >> 8, np = m & 255;
                size_t bh = (size_t)(b * 16 + h);
                u16 bv16 = f2bf(v);
                if (s == 0)      Qb[(bh * 256 + np) * 64 + d] = bv16;
                else if (s == 1) Kb[(bh * 256 + np) * 64 + d] = bv16;
                else             VTb[(bh * 64 + d) * 256 + np] = bv16;  // V transposed (B,H,D,N)
            }
}

// ---------- RoPE (in-place on Q and K, bf16) ----------
__global__ void rope_kernel(u16* __restrict__ Qb, u16* __restrict__ Kb) {
    u16* buf = (blockIdx.y == 0) ? Qb : Kb;
    int idx = blockIdx.x * blockDim.x + threadIdx.x;   // 8,388,608 pairs
    int dp = idx & 31;
    int n  = (idx >> 5) & 255;
    int bh = idx >> 13;
    int d = dp << 1;
    int j, tpos;
    if (d < 32) { j = d >> 1;        tpos = n >> 4; }
    else        { j = (d - 32) >> 1; tpos = n & 15; }
    // freq = 10000^(-j/16) = e^(-j*ln(1e4)/16)
    float freq = __expf(-0.5756462732485115f * (float)j);
    float ang = (float)tpos * freq;
    float sn, cs;
    __sincosf(ang, &sn, &cs);
    u16* p = buf + ((size_t)bh * 256 + n) * 64 + d;
    unsigned pr = *(unsigned*)p;
    float x0 = bf2f((u16)(pr & 0xffffu));
    float x1 = bf2f((u16)(pr >> 16));
    float y0 = x0 * cs - x1 * sn;
    float y1 = x1 * cs + x0 * sn;
    *(unsigned*)p = (unsigned)f2bf(y0) | ((unsigned)f2bf(y1) << 16);
}

// ---------- attention: one block per (b,h); 16 waves, each owns a 16-row Q strip ----------
__global__ __launch_bounds__(512) void attn_kernel(const u16* __restrict__ Qb, const u16* __restrict__ Kb,
                                                   const u16* __restrict__ VTb, u16* __restrict__ AT) {
    __shared__ __align__(16) u16 ps[16][16 * 32];   // per-wave P staging (16 KB)
    const int lane = threadIdx.x & 31;
    const int wid  = threadIdx.x >> 5;
    const int bh = blockIdx.x;
    const u16* Qp = Qb  + (size_t)bh * (256 * 64);
    const u16* Kp = Kb  + (size_t)bh * (256 * 64);
    const u16* Vt = VTb + (size_t)bh * (64 * 256);
    const int m0 = wid * 16;

    const f32x8 vz = {0.f, 0.f, 0.f, 0.f, 0.f, 0.f, 0.f, 0.f};
    f32x8 acc[16];
#pragma unroll
    for (int t = 0; t < 16; ++t) acc[t] = vz;

    // S = Q Kt : A = Q strip (16xD), B = K rows (contiguous d per lane)
#pragma unroll
    for (int d0 = 0; d0 < 64; d0 += 32) {
        bf16x16 aq = load_a16x32(Qp, m0, 64, d0, lane);
#pragma unroll
        for (int t = 0; t < 16; ++t) {
            bf16x16 bk = load_b32x16(Kp, t * 16, 64, d0, lane);
            acc[t] = WMMA_BF16(aq, bk, acc[t]);
        }
    }

    // softmax over 256 cols: per-register-row max/sum, xor-reduce inside each 16-lane half
    const float scale = 0.125f;                       // 1/sqrt(64)
    float rinv[8];
#pragma unroll
    for (int r = 0; r < 8; ++r) {
        float m = -3.0e38f;
#pragma unroll
        for (int t = 0; t < 16; ++t) m = fmaxf(m, acc[t][r]);
        m = fmaxf(m, SWZF(m, 0x041f));
        m = fmaxf(m, SWZF(m, 0x081f));
        m = fmaxf(m, SWZF(m, 0x101f));
        m = fmaxf(m, SWZF(m, 0x201f));
        float s = 0.f;
#pragma unroll
        for (int t = 0; t < 16; ++t) {
            float p = __expf((acc[t][r] - m) * scale);
            acc[t][r] = p;
            s += p;
        }
        s += SWZF(s, 0x041f);
        s += SWZF(s, 0x081f);
        s += SWZF(s, 0x101f);
        s += SWZF(s, 0x201f);
        rinv[r] = __frcp_rn(s);
    }
#pragma unroll
    for (int t = 0; t < 16; ++t)
#pragma unroll
        for (int r = 0; r < 8; ++r) acc[t][r] *= rinv[r];

    // O = P V : stage 16x32 P chunk through LDS (C-layout -> A-layout), B from V^T
    f32x8 o[4];
#pragma unroll
    for (int dt = 0; dt < 4; ++dt) o[dt] = vz;
    u16* myps = &ps[wid][0];
    const int c = lane & 15;
    const int rowadd = (lane < 16) ? 0 : 8;
#pragma unroll
    for (int jc = 0; jc < 8; ++jc) {
#pragma unroll
        for (int tt = 0; tt < 2; ++tt) {
            const int t = 2 * jc + tt;
#pragma unroll
            for (int r = 0; r < 8; ++r)
                myps[(r + rowadd) * 32 + tt * 16 + c] = f2bf(acc[t][r]);
        }
        // reload as A operand (row-major 16x32 in LDS)
        const u16* rp = myps + (lane & 15) * 32 + ((lane >> 4) << 3);
        u32x4 lo = *(const u32x4*)(rp);
        u32x4 hi = *(const u32x4*)(rp + 16);
        bf16x16 pa = __builtin_bit_cast(bf16x16, __builtin_shufflevector(lo, hi, 0, 1, 2, 3, 4, 5, 6, 7));
#pragma unroll
        for (int dt = 0; dt < 4; ++dt) {
            bf16x16 vb = load_b32x16(Vt, dt * 16, 256, jc * 32, lane);
            o[dt] = WMMA_BF16(pa, vb, o[dt]);
        }
    }

    // store to attention output, layout (B, N, H*D) bf16
    const int b = bh >> 4, h = bh & 15;
#pragma unroll
    for (int dt = 0; dt < 4; ++dt)
#pragma unroll
        for (int r = 0; r < 8; ++r) {
            int mrow = m0 + r + rowadd;
            int col = h * 64 + dt * 16 + c;
            AT[((size_t)(b * 256 + mrow)) * 1024 + col] = f2bf(o[dt][r]);
        }
}

// ---------- proj GEMM: (16384x1024) @ (1024x1024)^T + bias, fp32 out ----------
__global__ __launch_bounds__(256) void gemm_proj(const u16* __restrict__ A, const u16* __restrict__ W,
                                                 const float* __restrict__ bias, float* __restrict__ out) {
    const int lane  = threadIdx.x & 31;
    const int wid   = threadIdx.x >> 5;
    const int mbase = blockIdx.y * 128 + (wid >> 2) * 64;
    const int nbase = blockIdx.x * 128 + (wid & 3) * 32;

    const f32x8 vz = {0.f, 0.f, 0.f, 0.f, 0.f, 0.f, 0.f, 0.f};
    f32x8 acc[4][2];
#pragma unroll
    for (int i = 0; i < 4; ++i) { acc[i][0] = vz; acc[i][1] = vz; }

    for (int k0 = 0; k0 < 1024; k0 += 32) {
        if (k0 + 32 < 1024) {
            __builtin_prefetch(A + (size_t)(mbase + (lane & 15)) * 1024 + k0 + 32, 0, 1);
            __builtin_prefetch(W + (size_t)(nbase + (lane & 15)) * 1024 + k0 + 32, 0, 1);
        }
        bf16x16 av[4];
#pragma unroll
        for (int i = 0; i < 4; ++i) av[i] = load_a16x32(A, mbase + i * 16, 1024, k0, lane);
        bf16x16 bv[2];
#pragma unroll
        for (int j = 0; j < 2; ++j) bv[j] = load_b32x16(W, nbase + j * 16, 1024, k0, lane);
#pragma unroll
        for (int i = 0; i < 4; ++i)
#pragma unroll
            for (int j = 0; j < 2; ++j) acc[i][j] = WMMA_BF16(av[i], bv[j], acc[i][j]);
    }

    const int cc = lane & 15;
    const int rowadd = (lane < 16) ? 0 : 8;
#pragma unroll
    for (int i = 0; i < 4; ++i)
#pragma unroll
        for (int j = 0; j < 2; ++j)
#pragma unroll
            for (int r = 0; r < 8; ++r) {
                int m = mbase + i * 16 + r + rowadd;
                int n = nbase + j * 16 + cc;
                out[(size_t)m * 1024 + n] = acc[i][j][r] + bias[n];
            }
}

// ---------- launch ----------
extern "C" void kernel_launch(void* const* d_in, const int* in_sizes, int n_in,
                              void* d_out, int out_size, void* d_ws, size_t ws_size,
                              hipStream_t stream) {
    const float* x      = (const float*)d_in[0];   // (64,256,1024)
    const float* qkv_w  = (const float*)d_in[1];   // (3072,1024)
    const float* proj_w = (const float*)d_in[2];   // (1024,1024)
    const float* proj_b = (const float*)d_in[3];   // (1024,)
    float* out = (float*)d_out;                    // (64,256,1024) fp32

    char* ws = (char*)d_ws;
    const size_t SZ_X   = 33554432;  // 16384*1024*2
    const size_t SZ_WQ  = 6291456;   // 3072*1024*2
    const size_t SZ_WP  = 2097152;   // 1024*1024*2
    const size_t SZ_QKV = 33554432;  // 64*16*256*64*2
    u16* xb  = (u16*)(ws);
    u16* wq  = (u16*)(ws + SZ_X);
    u16* wp  = (u16*)(ws + SZ_X + SZ_WQ);
    u16* Qb  = (u16*)(ws + SZ_X + SZ_WQ + SZ_WP);
    u16* Kb  = (u16*)(ws + SZ_X + SZ_WQ + SZ_WP + SZ_QKV);
    u16* VTb = (u16*)(ws + SZ_X + SZ_WQ + SZ_WP + 2 * SZ_QKV);
    u16* AT  = (u16*)(ws + SZ_X + SZ_WQ + SZ_WP + 3 * SZ_QKV);

    // fp32 -> bf16
    cvt_kernel<<<16384, 256, 0, stream>>>(x,      xb, 4194304);
    cvt_kernel<<<3072,  256, 0, stream>>>(qkv_w,  wq, 786432);
    cvt_kernel<<<1024,  256, 0, stream>>>(proj_w, wp, 262144);

    // qkv projection + q/k/v(T) scatter
    gemm_qkv<<<dim3(24, 128), 256, 0, stream>>>(xb, wq, Qb, Kb, VTb);

    // RoPE on q and k
    rope_kernel<<<dim3(32768, 2), 256, 0, stream>>>(Qb, Kb);

    // attention
    attn_kernel<<<1024, 512, 0, stream>>>(Qb, Kb, VTb, AT);

    // output projection + bias
    gemm_proj<<<dim3(8, 128), 256, 0, stream>>>(AT, wp, proj_b, out);
}
